// TransformerRoPE_35141422416168
// MI455X (gfx1250) — matrix operand order repaired and checked by reading, not compiled, not run on hardware
//
#include <hip/hip_runtime.h>
#include <hip/hip_bf16.h>
#include <math.h>

typedef __bf16 bf16;
typedef __bf16 bf16x8  __attribute__((ext_vector_type(8)));
typedef __bf16 bf16x16 __attribute__((ext_vector_type(16)));
typedef float  f32x8   __attribute__((ext_vector_type(8)));
typedef unsigned int u32x4 __attribute__((ext_vector_type(4)));
typedef int          i32x4 __attribute__((ext_vector_type(4)));
typedef int          i32x8 __attribute__((ext_vector_type(8)));

#define LNUM 2
#define D    1024
#define H    16
#define DH   64
#define DFF  4096
#define BSZ  2
#define TSEQ 2048
#define OV   4096
#define MTOK (BSZ * TSEQ)   // 4096 tokens

// ---------------------------------------------------------------------------
// WMMA helpers (CDNA5 v_wmma_f32_16x16x32_bf16, wave32)
// ---------------------------------------------------------------------------
__device__ __forceinline__ f32x8 wmma_bf16(bf16x16 a, bf16x16 b, f32x8 c) {
  return __builtin_amdgcn_wmma_f32_16x16x32_bf16(false, a, false, b, (short)0,
                                                 c, false, false);
}

// A-fragment: 16x32 tile, row-major source with leading dim lda (elements).
__device__ __forceinline__ bf16x16 load_a_frag(const bf16* A, int lda, int lane) {
  int half = lane >> 4, r = lane & 15;
  const bf16* p = A + (size_t)r * lda + half * 16;
  bf16x8 lo = *(const bf16x8*)p;
  bf16x8 hi = *(const bf16x8*)(p + 8);
  bf16x16 f;
#pragma unroll
  for (int e = 0; e < 8; ++e) { f[e] = lo[e]; f[e + 8] = hi[e]; }
  return f;
}

// B-fragment: 32x16 tile from transposed storage Bt[N][K] (ldb = K).
__device__ __forceinline__ bf16x16 load_b_frag(const bf16* Bt, int ldb, int lane) {
  int half = lane >> 4, c = lane & 15;
  return *(const bf16x16*)(Bt + (size_t)c * ldb + half * 16);
}

// ---------------------------------------------------------------------------
// Tensor Data Mover: async 2-D tile load (rows x 32 bf16) global -> LDS.
// Builds D# groups per CDNA5 ISA.
// ---------------------------------------------------------------------------
__device__ __forceinline__ unsigned lds_off(const void* p) {
  // low 32 bits of the flat shared-aperture address == LDS byte offset
  return (unsigned)(unsigned long long)p;
}

__device__ __forceinline__ void tdm_load_2d(const bf16* gptr, unsigned ldsaddr,
                                            int rows, int k_stride) {
  unsigned long long ga = (unsigned long long)gptr;
  u32x4 g0;
  g0[0] = 1u;                                             // count=1, user D#
  g0[1] = ldsaddr;                                        // lds_addr (bytes)
  g0[2] = (unsigned)(ga & 0xFFFFFFFFu);                   // global_addr[31:0]
  g0[3] = (unsigned)((ga >> 32) & 0x1FFFFFFu) | (2u << 30); // ga[56:32] | type=2

  i32x8 g1;
  g1[0] = 1 << 16;                                        // data_size=1 (2B)
  g1[1] = (k_stride & 0xFFFF) << 16;                      // tensor_dim0[15:0]
  g1[2] = ((k_stride >> 16) & 0xFFFF) | ((rows & 0xFFFF) << 16); // dim0 hi | dim1 lo
  g1[3] = ((rows >> 16) & 0xFFFF) | (32 << 16);           // dim1 hi | tile_dim0=32
  g1[4] = rows & 0xFFFF;                                  // tile_dim1=rows, tile_dim2=0
  g1[5] = k_stride;                                       // tensor_dim0_stride[31:0]
  g1[6] = 0;                                              // stride hi | dim1_stride lo
  g1[7] = 0;

  i32x4 gz = {0, 0, 0, 0};
#if defined(__clang_major__) && (__clang_major__ >= 23)
  i32x8 gz8 = {0, 0, 0, 0, 0, 0, 0, 0};
  __builtin_amdgcn_tensor_load_to_lds(g0, g1, gz, gz, gz8, 0);
#else
  __builtin_amdgcn_tensor_load_to_lds(g0, g1, gz, gz, 0);
#endif
}

// ---------------------------------------------------------------------------
// GEMM: C[M,N] = A[M,K](bf16) @ Bt[N,K](bf16)^T + bias, optional GELU /
// residual-add, fp32 or bf16 output. Block = 8 waves -> 128x256 tile,
// wave tile 64x64 (4x4 WMMA accumulators). K panels staged to LDS by the
// Tensor Data Mover, double-buffered and overlapped with WMMA compute.
// ---------------------------------------------------------------------------
__global__ void __launch_bounds__(256)
gemm_wmma_kernel(const bf16* __restrict__ A, const bf16* __restrict__ Bt,
                 const float* __restrict__ bias, const float* __restrict__ residual,
                 float* __restrict__ out_f32, bf16* __restrict__ out_bf16,
                 int M, int N, int K, int do_gelu) {
  __shared__ bf16 sA[2][128 * 32];   // 2 x 8 KB
  __shared__ bf16 sB[2][256 * 32];   // 2 x 16 KB

  int lane = threadIdx.x & 31;
  int wave = threadIdx.x >> 5;
  int wm = wave >> 2, wn = wave & 3;
  int mblk = blockIdx.y * 128;
  int nblk = blockIdx.x * 256;

  f32x8 zero = {0.f, 0.f, 0.f, 0.f, 0.f, 0.f, 0.f, 0.f};
  f32x8 acc[4][4];
#pragma unroll
  for (int i = 0; i < 4; ++i)
#pragma unroll
    for (int j = 0; j < 4; ++j) acc[i][j] = zero;

  // ---- TDM preload of the first K panel ----
  if (wave == 0) {
    tdm_load_2d(A + (size_t)mblk * K, lds_off(&sA[0][0]), 128, K);
    tdm_load_2d(Bt + (size_t)nblk * K, lds_off(&sB[0][0]), 256, K);
    __builtin_amdgcn_s_wait_tensorcnt(0);
  }
  __syncthreads();

  int buf = 0;
  for (int k0 = 0; k0 < K; k0 += 32) {
    // kick off DMA for the next panel into the other buffer
    if (wave == 0 && k0 + 32 < K) {
      tdm_load_2d(A + (size_t)mblk * K + (k0 + 32), lds_off(&sA[buf ^ 1][0]), 128, K);
      tdm_load_2d(Bt + (size_t)nblk * K + (k0 + 32), lds_off(&sB[buf ^ 1][0]), 256, K);
    }
    // warm L2 for the panel after next (global_prefetch_b8)
    if (k0 + 64 < K) {
      __builtin_prefetch(A + (size_t)(mblk + (int)(threadIdx.x >> 1)) * K + k0 + 64, 0, 1);
      __builtin_prefetch(Bt + (size_t)(nblk + (int)threadIdx.x) * K + k0 + 64, 0, 1);
    }

    bf16x16 af[4], bfg[4];
#pragma unroll
    for (int i = 0; i < 4; ++i)
      af[i] = load_a_frag(&sA[buf][(wm * 64 + 16 * i) * 32], 32, lane);
#pragma unroll
    for (int j = 0; j < 4; ++j)
      bfg[j] = load_b_frag(&sB[buf][(wn * 64 + 16 * j) * 32], 32, lane);
#pragma unroll
    for (int i = 0; i < 4; ++i)
#pragma unroll
      for (int j = 0; j < 4; ++j)
        acc[i][j] = wmma_bf16(af[i], bfg[j], acc[i][j]);

    __syncthreads();   // all waves done reading buf
    if (wave == 0 && k0 + 32 < K) __builtin_amdgcn_s_wait_tensorcnt(0);
    __syncthreads();   // next buffer published
    buf ^= 1;
  }

  int half = lane >> 4, c = lane & 15;
#pragma unroll
  for (int i = 0; i < 4; ++i)
#pragma unroll
    for (int j = 0; j < 4; ++j)
#pragma unroll
      for (int r = 0; r < 8; ++r) {
        int row = mblk + wm * 64 + 16 * i + half * 8 + r;
        int col = nblk + wn * 64 + 16 * j + c;
        float v = acc[i][j][r];
        if (bias) v += bias[col];
        if (do_gelu) v = 0.5f * v * (1.0f + erff(v * 0.70710678118f));
        if (residual) v += residual[(size_t)row * N + col];
        if (out_f32) out_f32[(size_t)row * N + col] = v;
        else         out_bf16[(size_t)row * N + col] = (bf16)v;
      }
}

// ---------------------------------------------------------------------------
// Flash-style causal attention. One wave per (b, h, 32-row q tile).
// Q@K^T via WMMA, online softmax through LDS, P@V via WMMA.
// ---------------------------------------------------------------------------
__global__ void __launch_bounds__(32)
attn_kernel(const bf16* __restrict__ Qb, const bf16* __restrict__ Kb,
            const bf16* __restrict__ Vt, bf16* __restrict__ Ob) {
  __shared__ float s_s[32 * 32];
  __shared__ bf16  s_p[32 * 32];
  __shared__ float s_fac[32];

  int lane = threadIdx.x & 31;
  int bh = blockIdx.y, b = bh / H, h = bh % H;
  int qt0 = blockIdx.x * 32;
  int half = lane >> 4, c = lane & 15;

  const bf16* Qbase = Qb + (size_t)b * TSEQ * D + h * DH;
  const bf16* Kbase = Kb + (size_t)b * TSEQ * D + h * DH;
  const bf16* Vbase = Vt + (size_t)bh * DH * TSEQ;  // [DH][T]

  bf16x16 aq[2][2];
#pragma unroll
  for (int i = 0; i < 2; ++i)
#pragma unroll
    for (int kk = 0; kk < 2; ++kk)
      aq[i][kk] = load_a_frag(Qbase + (size_t)(qt0 + 16 * i) * D + kk * 32, D, lane);

  f32x8 zero = {0.f, 0.f, 0.f, 0.f, 0.f, 0.f, 0.f, 0.f};
  f32x8 oacc[2][4];
#pragma unroll
  for (int i = 0; i < 2; ++i)
#pragma unroll
    for (int j = 0; j < 4; ++j) oacc[i][j] = zero;

  float m_i = -3.0e38f, l_i = 0.f;  // per-lane row stats (row == lane)

  int nkt = blockIdx.x + 1;  // causal: only tiles with kb0 <= qt0+31
  for (int kt = 0; kt < nkt; ++kt) {
    int kb0 = kt * 32;
    f32x8 sac[2][2];
#pragma unroll
    for (int i = 0; i < 2; ++i)
#pragma unroll
      for (int j = 0; j < 2; ++j) sac[i][j] = zero;

#pragma unroll
    for (int j = 0; j < 2; ++j) {
      bf16x16 bk0 = load_b_frag(Kbase + (size_t)(kb0 + 16 * j) * D + 0,  D, lane);
      bf16x16 bk1 = load_b_frag(Kbase + (size_t)(kb0 + 16 * j) * D + 32, D, lane);
#pragma unroll
      for (int i = 0; i < 2; ++i) {
        sac[i][j] = wmma_bf16(aq[i][0], bk0, sac[i][j]);
        sac[i][j] = wmma_bf16(aq[i][1], bk1, sac[i][j]);
      }
    }

    // scale (1/sqrt(64)) + causal mask, spill to LDS
#pragma unroll
    for (int i = 0; i < 2; ++i)
#pragma unroll
      for (int j = 0; j < 2; ++j)
#pragma unroll
        for (int r = 0; r < 8; ++r) {
          int ql = 16 * i + half * 8 + r;
          int kl = 16 * j + c;
          float v = sac[i][j][r] * 0.125f;
          if (kb0 + kl > qt0 + ql) v = -3.0e38f;
          s_s[ql * 32 + kl] = v;
        }
    __syncthreads();

    // online softmax: lane owns row `lane`
    float tmax = -3.0e38f;
    for (int k = 0; k < 32; ++k) tmax = fmaxf(tmax, s_s[lane * 32 + k]);
    float m_new = fmaxf(m_i, tmax);
    float fac = __expf(m_i - m_new);
    float tsum = 0.f;
    for (int k = 0; k < 32; ++k) {
      float p = __expf(s_s[lane * 32 + k] - m_new);
      s_p[lane * 32 + k] = (bf16)p;
      tsum += p;
    }
    l_i = l_i * fac + tsum;
    m_i = m_new;
    s_fac[lane] = fac;
    __syncthreads();

    // rescale running output, then P @ V
#pragma unroll
    for (int i = 0; i < 2; ++i) {
      float fr[8];
#pragma unroll
      for (int r = 0; r < 8; ++r) fr[r] = s_fac[16 * i + half * 8 + r];
#pragma unroll
      for (int j = 0; j < 4; ++j)
#pragma unroll
        for (int r = 0; r < 8; ++r) oacc[i][j][r] *= fr[r];
    }

    bf16x16 pa[2];
#pragma unroll
    for (int i = 0; i < 2; ++i)
      pa[i] = load_a_frag(s_p + (size_t)16 * i * 32, 32, lane);
#pragma unroll
    for (int j = 0; j < 4; ++j) {
      bf16x16 vb = load_b_frag(Vbase + (size_t)16 * j * TSEQ + kb0, TSEQ, lane);
#pragma unroll
      for (int i = 0; i < 2; ++i)
        oacc[i][j] = wmma_bf16(pa[i], vb, oacc[i][j]);
    }
    __syncthreads();
  }

  // normalize by row sums and store bf16
  s_fac[lane] = 1.0f / l_i;
  __syncthreads();
#pragma unroll
  for (int i = 0; i < 2; ++i) {
    float fr[8];
#pragma unroll
    for (int r = 0; r < 8; ++r) fr[r] = s_fac[16 * i + half * 8 + r];
#pragma unroll
    for (int j = 0; j < 4; ++j)
#pragma unroll
      for (int r = 0; r < 8; ++r) {
        int row = qt0 + 16 * i + half * 8 + r;
        int col = 16 * j + c;
        Ob[((size_t)(b * TSEQ + row)) * D + h * DH + col] = (bf16)(oacc[i][j][r] * fr[r]);
      }
  }
}

// ---------------------------------------------------------------------------
// Elementwise / reduction kernels (fp32 residual stream)
// ---------------------------------------------------------------------------
__global__ void embed_kernel(const int* __restrict__ actions,
                             const int* __restrict__ obs,
                             const float* __restrict__ aemb,
                             const float* __restrict__ oemb,
                             const float* __restrict__ temb,
                             float* __restrict__ x) {
  size_t idx = (size_t)blockIdx.x * 256 + threadIdx.x;
  if (idx >= (size_t)MTOK * D) return;
  int m = (int)(idx / D), d = (int)(idx % D);
  x[idx] = aemb[(size_t)actions[m] * D + d] + temb[d] +
           oemb[(size_t)obs[m] * D + d] + temb[D + d];
}

__global__ void __launch_bounds__(256)
ln_kernel(const float* __restrict__ x, const float* __restrict__ g,
          const float* __restrict__ b, bf16* __restrict__ out) {
  __shared__ float red[256];
  int row = blockIdx.x;
  const float* xr = x + (size_t)row * D;
  int tid = threadIdx.x;

  float s = 0.f;
  for (int d = tid; d < D; d += 256) s += xr[d];
  red[tid] = s; __syncthreads();
  for (int off = 128; off > 0; off >>= 1) {
    if (tid < off) red[tid] += red[tid + off];
    __syncthreads();
  }
  float mean = red[0] * (1.0f / D);
  __syncthreads();

  float v = 0.f;
  for (int d = tid; d < D; d += 256) { float t = xr[d] - mean; v += t * t; }
  red[tid] = v; __syncthreads();
  for (int off = 128; off > 0; off >>= 1) {
    if (tid < off) red[tid] += red[tid + off];
    __syncthreads();
  }
  float rstd = rsqrtf(red[0] * (1.0f / D) + 1e-5f);

  for (int d = tid; d < D; d += 256)
    out[(size_t)row * D + d] = (bf16)((xr[d] - mean) * rstd * g[d] + b[d]);
}

// fp32 W[K,N] -> bf16 Wt[N,K]
__global__ void transpose_bf16_kernel(const float* __restrict__ W,
                                      bf16* __restrict__ Wt,
                                      int Krows, int Ncols) {
  size_t idx = (size_t)blockIdx.x * 256 + threadIdx.x;
  if (idx >= (size_t)Krows * Ncols) return;
  int k = (int)(idx / Ncols), n = (int)(idx % Ncols);
  Wt[(size_t)n * Krows + k] = (bf16)W[idx];
}

__global__ void rope_kernel(const float* __restrict__ Qf,
                            const float* __restrict__ Kf,
                            bf16* __restrict__ Qb, bf16* __restrict__ Kb) {
  const int DH2 = DH / 2;
  size_t idx = (size_t)blockIdx.x * 256 + threadIdx.x;
  if (idx >= (size_t)MTOK * H * DH2) return;
  int i = (int)(idx % DH2);
  int tmp = (int)(idx / DH2);
  int h = tmp % H, m = tmp / H;
  int t = m % TSEQ;
  float freq = (float)t * __powf(10000.0f, -2.0f * (float)i / (float)DH);
  float cs = __cosf(freq), sn = __sinf(freq);
  size_t base = (size_t)m * D + h * DH + 2 * i;
  float q1 = Qf[base], q2 = Qf[base + 1];
  Qb[base]     = (bf16)(q1 * cs - q2 * sn);
  Qb[base + 1] = (bf16)(q1 * sn + q2 * cs);
  float k1 = Kf[base], k2 = Kf[base + 1];
  Kb[base]     = (bf16)(k1 * cs - k2 * sn);
  Kb[base + 1] = (bf16)(k1 * sn + k2 * cs);
}

// V[m, h*DH+d] (fp32) -> Vt[(b*H+h)*DH + d][t] (bf16)
__global__ void vtrans_kernel(const float* __restrict__ Vf, bf16* __restrict__ Vt) {
  size_t idx = (size_t)blockIdx.x * 256 + threadIdx.x;
  if (idx >= (size_t)MTOK * D) return;
  int m = (int)(idx / D), dd = (int)(idx % D);
  int h = dd / DH, d = dd % DH;
  int b = m / TSEQ, t = m % TSEQ;
  Vt[(((size_t)(b * H + h)) * DH + d) * TSEQ + t] = (bf16)Vf[idx];
}

// ---------------------------------------------------------------------------
// Host orchestration
// ---------------------------------------------------------------------------
extern "C" void kernel_launch(void* const* d_in, const int* in_sizes, int n_in,
                              void* d_out, int out_size, void* d_ws, size_t ws_size,
                              hipStream_t stream) {
  const int*   actions = (const int*)d_in[0];
  const int*   observ  = (const int*)d_in[1];
  const float* aemb  = (const float*)d_in[2];
  const float* oemb  = (const float*)d_in[3];
  const float* temb  = (const float*)d_in[4];
  const float* Wq    = (const float*)d_in[5];
  const float* bq    = (const float*)d_in[6];
  const float* Wk    = (const float*)d_in[7];
  const float* bk    = (const float*)d_in[8];
  const float* Wv    = (const float*)d_in[9];
  const float* bv    = (const float*)d_in[10];
  const float* Wo    = (const float*)d_in[11];
  const float* bo    = (const float*)d_in[12];
  const float* ln1_g = (const float*)d_in[13];
  const float* ln1_b = (const float*)d_in[14];
  const float* ln2_g = (const float*)d_in[15];
  const float* ln2_b = (const float*)d_in[16];
  const float* W1    = (const float*)d_in[17];
  const float* b1    = (const float*)d_in[18];
  const float* W2    = (const float*)d_in[19];
  const float* b2    = (const float*)d_in[20];
  const float* outg  = (const float*)d_in[21];
  const float* outb  = (const float*)d_in[22];
  const float* Wout  = (const float*)d_in[23];
  const float* bout  = (const float*)d_in[24];
  float* out = (float*)d_out;

  char* ws = (char*)d_ws;
  size_t off = 0;
  auto alloc = [&](size_t bytes) -> char* {
    char* p = ws + off;
    off += (bytes + 255) & ~(size_t)255;
    return p;
  };

  // bf16 transposed weights (persistent through the launch)
  bf16* WqT   = (bf16*)alloc((size_t)LNUM * D * D * 2);
  bf16* WkT   = (bf16*)alloc((size_t)LNUM * D * D * 2);
  bf16* WvT   = (bf16*)alloc((size_t)LNUM * D * D * 2);
  bf16* WoT   = (bf16*)alloc((size_t)LNUM * D * D * 2);
  bf16* W1T   = (bf16*)alloc((size_t)LNUM * DFF * D * 2);  // [DFF, D]
  bf16* W2T   = (bf16*)alloc((size_t)LNUM * D * DFF * 2);  // [D, DFF]
  bf16* WoutT = (bf16*)alloc((size_t)OV * D * 2);          // [OV, D]
  // activations
  float* x   = (float*)alloc((size_t)MTOK * D * 4);
  bf16*  hbf = (bf16*)alloc((size_t)MTOK * D * 2);
  bf16*  Qbf = (bf16*)alloc((size_t)MTOK * D * 2);
  bf16*  Kbf = (bf16*)alloc((size_t)MTOK * D * 2);
  bf16*  Vt  = (bf16*)alloc((size_t)MTOK * D * 2);
  bf16*  Obf = (bf16*)alloc((size_t)MTOK * D * 2);
  // fp32 QKV scratch; FFN hidden (bf16, 32MB) aliases this 48MB region
  float* Qf  = (float*)alloc((size_t)MTOK * D * 4 * 3);
  float* Kf  = Qf + (size_t)MTOK * D;
  float* Vf  = Kf + (size_t)MTOK * D;
  bf16*  ffbf = (bf16*)Qf;  // [MTOK, DFF] bf16, used after QKV scratch is dead

  (void)in_sizes; (void)n_in; (void)out_size; (void)ws_size;

  auto g1 = [](size_t n) { return dim3((unsigned)((n + 255) / 256)); };

  // ---- weight convert + transpose (bf16) ----
  for (int l = 0; l < LNUM; ++l) {
    size_t wo = (size_t)l * D * D;
    transpose_bf16_kernel<<<g1((size_t)D * D), 256, 0, stream>>>(Wq + wo, WqT + wo, D, D);
    transpose_bf16_kernel<<<g1((size_t)D * D), 256, 0, stream>>>(Wk + wo, WkT + wo, D, D);
    transpose_bf16_kernel<<<g1((size_t)D * D), 256, 0, stream>>>(Wv + wo, WvT + wo, D, D);
    transpose_bf16_kernel<<<g1((size_t)D * D), 256, 0, stream>>>(Wo + wo, WoT + wo, D, D);
    transpose_bf16_kernel<<<g1((size_t)D * DFF), 256, 0, stream>>>(
        W1 + (size_t)l * D * DFF, W1T + (size_t)l * DFF * D, D, DFF);
    transpose_bf16_kernel<<<g1((size_t)DFF * D), 256, 0, stream>>>(
        W2 + (size_t)l * DFF * D, W2T + (size_t)l * D * DFF, DFF, D);
  }
  transpose_bf16_kernel<<<g1((size_t)D * OV), 256, 0, stream>>>(Wout, WoutT, D, OV);

  // ---- embedding ----
  embed_kernel<<<g1((size_t)MTOK * D), 256, 0, stream>>>(actions, observ, aemb, oemb, temb, x);

  // ---- transformer layers ----
  for (int l = 0; l < LNUM; ++l) {
    size_t wo = (size_t)l * D * D;
    ln_kernel<<<MTOK, 256, 0, stream>>>(x, ln1_g + (size_t)l * D, ln1_b + (size_t)l * D, hbf);

    dim3 gD(D / 256, MTOK / 128);
    gemm_wmma_kernel<<<gD, 256, 0, stream>>>(hbf, WqT + wo, bq + (size_t)l * D, nullptr,
                                             Qf, nullptr, MTOK, D, D, 0);
    gemm_wmma_kernel<<<gD, 256, 0, stream>>>(hbf, WkT + wo, bk + (size_t)l * D, nullptr,
                                             Kf, nullptr, MTOK, D, D, 0);
    gemm_wmma_kernel<<<gD, 256, 0, stream>>>(hbf, WvT + wo, bv + (size_t)l * D, nullptr,
                                             Vf, nullptr, MTOK, D, D, 0);

    rope_kernel<<<g1((size_t)MTOK * H * (DH / 2)), 256, 0, stream>>>(Qf, Kf, Qbf, Kbf);
    vtrans_kernel<<<g1((size_t)MTOK * D), 256, 0, stream>>>(Vf, Vt);

    attn_kernel<<<dim3(TSEQ / 32, BSZ * H), 32, 0, stream>>>(Qbf, Kbf, Vt, Obf);

    // x = x + O @ Wo + bo   (residual fused in epilogue)
    gemm_wmma_kernel<<<gD, 256, 0, stream>>>(Obf, WoT + wo, bo + (size_t)l * D, x,
                                             x, nullptr, MTOK, D, D, 0);

    ln_kernel<<<MTOK, 256, 0, stream>>>(x, ln2_g + (size_t)l * D, ln2_b + (size_t)l * D, hbf);

    // ff hidden = gelu(h @ W1 + b1) -> bf16
    gemm_wmma_kernel<<<dim3(DFF / 256, MTOK / 128), 256, 0, stream>>>(
        hbf, W1T + (size_t)l * DFF * D, b1 + (size_t)l * DFF, nullptr,
        nullptr, ffbf, MTOK, DFF, D, 1);
    // x = x + ff @ W2 + b2
    gemm_wmma_kernel<<<gD, 256, 0, stream>>>(
        ffbf, W2T + (size_t)l * D * DFF, b2 + (size_t)l * D, x,
        x, nullptr, MTOK, D, DFF, 0);
  }

  // ---- final LN + logits ----
  ln_kernel<<<MTOK, 256, 0, stream>>>(x, outg, outb, hbf);
  gemm_wmma_kernel<<<dim3(OV / 256, MTOK / 128), 256, 0, stream>>>(
      hbf, WoutT, bout, nullptr, out, nullptr, MTOK, OV, D, 0);
}
